// PamCatAttention_54425825575000
// MI455X (gfx1250) — matrix operand order repaired and checked by reading, not compiled, not run on hardware
//
#include <hip/hip_runtime.h>
#include <hip/hip_bf16.h>
#include <math.h>

typedef __bf16 bf16;
typedef __attribute__((ext_vector_type(16))) __bf16 v16bf;
typedef __attribute__((ext_vector_type(8)))  float  v8f;

#define WMMA_BF16(a, b, c) \
  __builtin_amdgcn_wmma_f32_16x16x32_bf16(false, (a), false, (b), (short)0, (c), false, false)

// ---------------- constants (from reference setup) ----------------
#define BB 32
#define NN 1024
#define CC 768
#define MM 64
#define HH 12
#define DD 64

// ---------------- WMMA fragment loaders (ISA 7.12.2 layouts) ----------------
// A (16x32 bf16): lane L holds row (L&15); lanes 0-15: K=0..7 then 16..23,
// lanes 16-31: K=8..15 then 24..31.
static __device__ inline v16bf load_a_frag(const bf16* p, int ld, int lane) {
  const int r = lane & 15, half = lane >> 4;
  const bf16* base = p + (size_t)r * ld + half * 8;
  v16bf a;
#pragma unroll
  for (int i = 0; i < 8; ++i) a[i] = base[i];
#pragma unroll
  for (int i = 0; i < 8; ++i) a[8 + i] = base[16 + i];
  return a;
}

// B (32x16 bf16) where B[k][n] = W[n][k] (W row-major, stride ld), p at (col0,k0):
// lane L = column (L&15); lanes 0-15: K=0..7 then 16..23, lanes 16-31: K=8..15 then 24..31.
static __device__ inline v16bf load_bT_frag(const bf16* p, int ld, int lane) {
  const int n = lane & 15, half = lane >> 4;
  const bf16* base = p + (size_t)n * ld + half * 8;
  v16bf b;
#pragma unroll
  for (int i = 0; i < 8; ++i) b[i] = base[i];
#pragma unroll
  for (int i = 0; i < 8; ++i) b[8 + i] = base[16 + i];
  return b;
}

// ---------------- f32 -> bf16 convert ----------------
__global__ void cvt_f32_bf16(const float* __restrict__ in, bf16* __restrict__ out, int n) {
  int i = blockIdx.x * blockDim.x + threadIdx.x;
  if (i < n) out[i] = (bf16)in[i];
}

// ---------------- x (B,N,C) f32 -> xT (B,C,N) bf16, LDS-tiled ----------------
__global__ void __launch_bounds__(256) transpose_x(const float* __restrict__ x,
                                                   bf16* __restrict__ xT) {
  __shared__ bf16 t[32][33];
  int blk = blockIdx.x;
  const int ct = blk % (CC / 32); blk /= (CC / 32);
  const int nt = blk % (NN / 32);
  const int b  = blk / (NN / 32);
  const int tx = threadIdx.x & 31, ty = threadIdx.x >> 5;  // ty: 0..7
  const float* xb = x + (size_t)b * NN * CC;
#pragma unroll
  for (int i = 0; i < 4; ++i) {
    int n = nt * 32 + ty + i * 8;
    t[ty + i * 8][tx] = (bf16)xb[(size_t)n * CC + ct * 32 + tx];
  }
  __syncthreads();
  bf16* xTb = xT + (size_t)b * CC * NN;
#pragma unroll
  for (int i = 0; i < 4; ++i) {
    int c = ct * 32 + ty + i * 8;
    xTb[(size_t)c * NN + nt * 32 + tx] = t[tx][ty + i * 8];
  }
}

// ---------------- generic GEMM: out = in(R,K) @ W(S,K)^T + bias ----------------
// 64x64 output tile per wave: 16 WMMAs per 32-deep k-step.
// OUT_MODE: 0 = f32 row-major, 1 = bf16 row-major,
//           2 = bf16 transposed per batch: out[(b*S + col)*64 + (row&63)]  (for vT)
template <int OUT_MODE>
__global__ void __launch_bounds__(128) gemm_wt(const bf16* __restrict__ in,
                                               const bf16* __restrict__ W,
                                               const float* __restrict__ bias,
                                               void* __restrict__ outp,
                                               int S, int K) {
  const int lane = threadIdx.x & 31;
  const int wave = blockIdx.x * (blockDim.x >> 5) + (threadIdx.x >> 5);
  const int ctiles = S >> 6;
  const int rt0 = (wave / ctiles) << 6;
  const int ct0 = (wave % ctiles) << 6;

  v8f acc[4][4] = {};
  for (int k0 = 0; k0 < K; k0 += 32) {
    v16bf b0 = load_bT_frag(W + (size_t)ct0 * K + k0, K, lane);
    v16bf b1 = load_bT_frag(W + (size_t)(ct0 + 16) * K + k0, K, lane);
    v16bf b2 = load_bT_frag(W + (size_t)(ct0 + 32) * K + k0, K, lane);
    v16bf b3 = load_bT_frag(W + (size_t)(ct0 + 48) * K + k0, K, lane);
#pragma unroll
    for (int rt = 0; rt < 4; ++rt) {
      v16bf a = load_a_frag(in + (size_t)(rt0 + rt * 16) * K + k0, K, lane);
      acc[rt][0] = WMMA_BF16(a, b0, acc[rt][0]);
      acc[rt][1] = WMMA_BF16(a, b1, acc[rt][1]);
      acc[rt][2] = WMMA_BF16(a, b2, acc[rt][2]);
      acc[rt][3] = WMMA_BF16(a, b3, acc[rt][3]);
    }
  }
  const int half = lane >> 4, cn = lane & 15;
#pragma unroll
  for (int rt = 0; rt < 4; ++rt)
#pragma unroll
    for (int ct = 0; ct < 4; ++ct)
#pragma unroll
      for (int r = 0; r < 8; ++r) {
        int row = rt0 + rt * 16 + r + half * 8;
        int col = ct0 + ct * 16 + cn;
        float v = acc[rt][ct][r] + bias[col];
        if (OUT_MODE == 0) {
          ((float*)outp)[(size_t)row * S + col] = v;
        } else if (OUT_MODE == 1) {
          ((bf16*)outp)[(size_t)row * S + col] = (bf16)v;
        } else {
          int bidx = row >> 6, m = row & 63;  // rows per batch = MM = 64
          ((bf16*)outp)[((size_t)bidx * S + col) * MM + m] = (bf16)v;
        }
      }
}

// ---------------- cluster scores: c[b,m,n] = sigmoid(Wc.x^T + bc)/N ----------------
__global__ void __launch_bounds__(128) cluster_scores(const bf16* __restrict__ xbf,
                                                      const bf16* __restrict__ Wc,
                                                      const float* __restrict__ bc,
                                                      bf16* __restrict__ cbf) {
  const int lane = threadIdx.x & 31;
  const int wave = blockIdx.x * (blockDim.x >> 5) + (threadIdx.x >> 5);
  const int ntt = NN / 32;                  // 32 col tiles
  const int tilesPerB = (MM / 32) * ntt;    // 64
  const int b = wave / tilesPerB;
  const int rem = wave % tilesPerB;
  const int m0 = (rem / ntt) << 5;
  const int n0 = (rem % ntt) << 5;
  const bf16* xb = xbf + (size_t)b * NN * CC;

  v8f acc[2][2] = {};
  for (int k0 = 0; k0 < CC; k0 += 32) {
    v16bf a0 = load_a_frag(Wc + (size_t)m0 * CC + k0, CC, lane);
    v16bf a1 = load_a_frag(Wc + (size_t)(m0 + 16) * CC + k0, CC, lane);
    v16bf b0 = load_bT_frag(xb + (size_t)n0 * CC + k0, CC, lane);
    v16bf b1 = load_bT_frag(xb + (size_t)(n0 + 16) * CC + k0, CC, lane);
    acc[0][0] = WMMA_BF16(a0, b0, acc[0][0]);
    acc[0][1] = WMMA_BF16(a0, b1, acc[0][1]);
    acc[1][0] = WMMA_BF16(a1, b0, acc[1][0]);
    acc[1][1] = WMMA_BF16(a1, b1, acc[1][1]);
  }
  const int half = lane >> 4, cn = lane & 15;
  const float invN = 1.0f / (float)NN;
#pragma unroll
  for (int rt = 0; rt < 2; ++rt)
#pragma unroll
    for (int ct = 0; ct < 2; ++ct)
#pragma unroll
      for (int r = 0; r < 8; ++r) {
        int m = m0 + rt * 16 + r + half * 8;
        int n = n0 + ct * 16 + cn;
        float v = acc[rt][ct][r] + bc[m];
        v = invN / (1.0f + expf(-v));
        cbf[((size_t)b * MM + m) * NN + n] = (bf16)v;
      }
}

// ---------------- cluster aggregate: z[b] = c[b] @ x[b]  (fp32 out) ----------------
// uses xT (B,C,N) so B-fragments are contiguous
__global__ void __launch_bounds__(128) cluster_agg(const bf16* __restrict__ cbf,
                                                   const bf16* __restrict__ xT,
                                                   float* __restrict__ zf) {
  const int lane = threadIdx.x & 31;
  const int wave = blockIdx.x * (blockDim.x >> 5) + (threadIdx.x >> 5);
  const int ctt = CC / 32;                  // 24
  const int tilesPerB = (MM / 32) * ctt;    // 48
  const int b = wave / tilesPerB;
  const int rem = wave % tilesPerB;
  const int m0 = (rem / ctt) << 5;
  const int c0 = (rem % ctt) << 5;
  const bf16* cb  = cbf + (size_t)b * MM * NN;
  const bf16* xTb = xT + (size_t)b * CC * NN;

  v8f acc[2][2] = {};
  for (int k0 = 0; k0 < NN; k0 += 32) {
    v16bf a0 = load_a_frag(cb + (size_t)m0 * NN + k0, NN, lane);
    v16bf a1 = load_a_frag(cb + (size_t)(m0 + 16) * NN + k0, NN, lane);
    v16bf b0 = load_bT_frag(xTb + (size_t)c0 * NN + k0, NN, lane);
    v16bf b1 = load_bT_frag(xTb + (size_t)(c0 + 16) * NN + k0, NN, lane);
    acc[0][0] = WMMA_BF16(a0, b0, acc[0][0]);
    acc[0][1] = WMMA_BF16(a0, b1, acc[0][1]);
    acc[1][0] = WMMA_BF16(a1, b0, acc[1][0]);
    acc[1][1] = WMMA_BF16(a1, b1, acc[1][1]);
  }
  const int half = lane >> 4, cn = lane & 15;
#pragma unroll
  for (int rt = 0; rt < 2; ++rt)
#pragma unroll
    for (int ct = 0; ct < 2; ++ct)
#pragma unroll
      for (int r = 0; r < 8; ++r) {
        int m = m0 + rt * 16 + r + half * 8;
        int c = c0 + ct * 16 + cn;
        zf[((size_t)b * MM + m) * CC + c] = acc[rt][ct][r];
      }
}

// ---------------- L2-normalize z rows -> bf16 ----------------
__global__ void __launch_bounds__(128) znorm(const float* __restrict__ zf,
                                             bf16* __restrict__ zbf) {
  const int lane = threadIdx.x & 31;
  const int row = blockIdx.x * (blockDim.x >> 5) + (threadIdx.x >> 5);  // B*M rows
  const float* p = zf + (size_t)row * CC;
  float ss = 0.0f;
  for (int i = lane; i < CC; i += 32) {
    float v = p[i];
    ss += v * v;
  }
#pragma unroll
  for (int m = 16; m; m >>= 1) ss += __shfl_xor(ss, m, 32);
  float inv = 1.0f / fmaxf(sqrtf(ss), 1e-12f);
  bf16* o = zbf + (size_t)row * CC;
  for (int i = lane; i < CC; i += 32) o[i] = (bf16)(p[i] * inv);
}

// ---------------- fused attention: per (b,h,32-row tile) ----------------
// q (B,N,C) bf16, k (B,M,C) bf16, vT (B,C,M) bf16 -> o (B,N,C) bf16
__global__ void __launch_bounds__(128) attention(const bf16* __restrict__ q,
                                                 const bf16* __restrict__ kk,
                                                 const bf16* __restrict__ vT,
                                                 bf16* __restrict__ o) {
  __shared__ bf16 Pls[4][32 * 64];
  const int lane = threadIdx.x & 31;
  const int wib = threadIdx.x >> 5;
  const int t = blockIdx.x * 4 + wib;
  const int nt = t & 31;        // N/32 = 32 tiles
  const int bh = t >> 5;
  const int h = bh % HH;
  const int b = bh / HH;

  const size_t qoff = ((size_t)b * NN + (size_t)nt * 32) * CC + h * DD;
  const size_t koff = (size_t)b * MM * CC + h * DD;

  // scores S(32 x 64) = q_tile(32,64) @ k^T
  v8f s[2][4] = {};
  for (int k0 = 0; k0 < DD; k0 += 32) {
    v16bf a0 = load_a_frag(q + qoff + k0, CC, lane);
    v16bf a1 = load_a_frag(q + qoff + (size_t)16 * CC + k0, CC, lane);
#pragma unroll
    for (int ct = 0; ct < 4; ++ct) {
      v16bf bfr = load_bT_frag(kk + koff + (size_t)(ct * 16) * CC + k0, CC, lane);
      s[0][ct] = WMMA_BF16(a0, bfr, s[0][ct]);
      s[1][ct] = WMMA_BF16(a1, bfr, s[1][ct]);
    }
  }

  // softmax over m (64 cols), scale = D^-0.5 = 0.125
  const float scale = 0.125f;
  const int half = lane >> 4, cn = lane & 15;
  bf16* P = Pls[wib];
#pragma unroll
  for (int rt = 0; rt < 2; ++rt) {
#pragma unroll
    for (int r = 0; r < 8; ++r) {
      float v0 = s[rt][0][r] * scale;
      float v1 = s[rt][1][r] * scale;
      float v2 = s[rt][2][r] * scale;
      float v3 = s[rt][3][r] * scale;
      float mx = fmaxf(fmaxf(v0, v1), fmaxf(v2, v3));
#pragma unroll
      for (int m = 8; m; m >>= 1) mx = fmaxf(mx, __shfl_xor(mx, m, 32));
      float e0 = expf(v0 - mx), e1 = expf(v1 - mx);
      float e2 = expf(v2 - mx), e3 = expf(v3 - mx);
      float sum = e0 + e1 + e2 + e3;
#pragma unroll
      for (int m = 8; m; m >>= 1) sum += __shfl_xor(sum, m, 32);
      float inv = 1.0f / sum;
      int row = rt * 16 + r + half * 8;
      P[row * 64 + cn + 0]  = (bf16)(e0 * inv);
      P[row * 64 + cn + 16] = (bf16)(e1 * inv);
      P[row * 64 + cn + 32] = (bf16)(e2 * inv);
      P[row * 64 + cn + 48] = (bf16)(e3 * inv);
    }
  }
  __syncthreads();

  // o(32 x 64) = P(32,64) @ v(64,64); B[k][d] = v[k][h*64+d] = vT[h*64+d][k]
  const size_t vtoff = ((size_t)b * CC + (size_t)h * DD) * MM;
  v8f oa[2][4] = {};
  for (int k0 = 0; k0 < MM; k0 += 32) {
    v16bf a0 = load_a_frag(P + k0, 64, lane);
    v16bf a1 = load_a_frag(P + 16 * 64 + k0, 64, lane);
#pragma unroll
    for (int ct = 0; ct < 4; ++ct) {
      v16bf bb = load_bT_frag(vT + vtoff + (size_t)(ct * 16) * MM + k0, MM, lane);
      oa[0][ct] = WMMA_BF16(a0, bb, oa[0][ct]);
      oa[1][ct] = WMMA_BF16(a1, bb, oa[1][ct]);
    }
  }

  const size_t obase = ((size_t)b * NN + (size_t)nt * 32) * CC + h * DD;
#pragma unroll
  for (int rt = 0; rt < 2; ++rt)
#pragma unroll
    for (int ct = 0; ct < 4; ++ct)
#pragma unroll
      for (int r = 0; r < 8; ++r) {
        int n = rt * 16 + r + half * 8;
        int d = ct * 16 + cn;
        o[obase + (size_t)n * CC + d] = (bf16)oa[rt][ct][r];
      }
}

// ---------------- host launcher ----------------
extern "C" void kernel_launch(void* const* d_in, const int* in_sizes, int n_in,
                              void* d_out, int out_size, void* d_ws, size_t ws_size,
                              hipStream_t stream) {
  (void)in_sizes; (void)n_in; (void)out_size; (void)ws_size;
  const float* x  = (const float*)d_in[0];
  const float* Wc = (const float*)d_in[1];
  const float* bc = (const float*)d_in[2];
  const float* Wq = (const float*)d_in[3];
  const float* bq = (const float*)d_in[4];
  const float* Wk = (const float*)d_in[5];
  const float* bk = (const float*)d_in[6];
  const float* Wv = (const float*)d_in[7];
  const float* bv = (const float*)d_in[8];
  const float* Wp = (const float*)d_in[9];
  const float* bp = (const float*)d_in[10];

  char* ws = (char*)d_ws;
  size_t off = 0;
  auto alloc = [&](size_t bytes) -> void* {
    void* p = ws + off;
    off += (bytes + 255) & ~(size_t)255;
    return p;
  };
  bf16* xbf = (bf16*)alloc((size_t)BB * NN * CC * 2);
  bf16* xT  = (bf16*)alloc((size_t)BB * CC * NN * 2);
  bf16* Wcb = (bf16*)alloc((size_t)MM * CC * 2);
  bf16* Wqb = (bf16*)alloc((size_t)CC * CC * 2);
  bf16* Wkb = (bf16*)alloc((size_t)CC * CC * 2);
  bf16* Wvb = (bf16*)alloc((size_t)CC * CC * 2);
  bf16* Wpb = (bf16*)alloc((size_t)CC * CC * 2);
  bf16* cbf = (bf16*)alloc((size_t)BB * MM * NN * 2);
  float* zf = (float*)alloc((size_t)BB * MM * CC * 4);
  bf16* zbf = (bf16*)alloc((size_t)BB * MM * CC * 2);
  bf16* qbf = (bf16*)alloc((size_t)BB * NN * CC * 2);
  bf16* kbf = (bf16*)alloc((size_t)BB * MM * CC * 2);
  bf16* vTb = (bf16*)alloc((size_t)BB * CC * MM * 2);
  bf16* obf = (bf16*)alloc((size_t)BB * NN * CC * 2);

  // converts + transpose
  {
    int n = BB * NN * CC;
    cvt_f32_bf16<<<(n + 255) / 256, 256, 0, stream>>>(x, xbf, n);
    transpose_x<<<BB * (NN / 32) * (CC / 32), 256, 0, stream>>>(x, xT);
    n = MM * CC;
    cvt_f32_bf16<<<(n + 255) / 256, 256, 0, stream>>>(Wc, Wcb, n);
    n = CC * CC;
    cvt_f32_bf16<<<(n + 255) / 256, 256, 0, stream>>>(Wq, Wqb, n);
    cvt_f32_bf16<<<(n + 255) / 256, 256, 0, stream>>>(Wk, Wkb, n);
    cvt_f32_bf16<<<(n + 255) / 256, 256, 0, stream>>>(Wv, Wvb, n);
    cvt_f32_bf16<<<(n + 255) / 256, 256, 0, stream>>>(Wp, Wpb, n);
  }

  // cluster path
  cluster_scores<<<512, 128, 0, stream>>>(xbf, Wcb, bc, cbf);   // 2048 tiles (32x32)
  cluster_agg<<<384, 128, 0, stream>>>(cbf, xT, zf);            // 1536 tiles (32x32)
  znorm<<<512, 128, 0, stream>>>(zf, zbf);                      // 2048 rows

  // projections (64x64 tiles/wave)
  gemm_wt<1><<<1536, 128, 0, stream>>>(xbf, Wqb, bq, qbf, CC, CC);  // q: 6144 tiles
  gemm_wt<1><<<96, 128, 0, stream>>>(zbf, Wkb, bk, kbf, CC, CC);    // k: 384 tiles
  gemm_wt<2><<<96, 128, 0, stream>>>(zbf, Wvb, bv, vTb, CC, CC);    // v -> vT: 384 tiles

  // fused attention
  attention<<<3072, 128, 0, stream>>>(qbf, kbf, vTb, obf);      // 12288 tiles

  // output projection (fp32 out)
  gemm_wt<0><<<1536, 128, 0, stream>>>(obf, Wpb, bp, d_out, CC, CC);
}
